// OscillatoryDPLRSSM_62302795596451
// MI455X (gfx1250) — matrix operand order repaired and checked by reading, not compiled, not run on hardware
//
#include <hip/hip_runtime.h>

// ---------------------------------------------------------------------------
// Oscillatory DPLR SSM on MI455X (gfx1250, wave32, WMMA).
//
// Pipeline:
//   K1 setup (1 wg):   discretize (complex Gauss-Jordan) -> A_d, B_d;
//                      build fused step matrix G (256x256, bf16):
//                        [Z'; Y] = G @ [Z; U],  Z=[Re x; Im x]
//                      A_L = A_d^64 (repeated squaring) -> M_L (128x128 f32)
//   K2 pass1 (128 wg): per-chunk local recurrence (zero init), 64 steps of a
//                      128x256 @ 256x32 bf16 WMMA GEMM per wg; store chunk
//                      end-states E.
//   K3 carry (32 wg):  sequential carry scan over 128 chunks per batch:
//                      S[c+1] = M_L S[c] + E[c]   (tiny)
//   K4 pass3 (128 wg): recurrence again with true init S, one fused
//                      256x256 @ 256x32 WMMA GEMM per step emits Z' and Y.
//
// ~40 GFLOP vs ~384 MB HBM traffic -> bandwidth bound; bf16 WMMA with f32
// accumulation keeps the matrix work on the matrix pipes.
// Workspace: ~4.7 MB.
// ---------------------------------------------------------------------------

typedef __attribute__((ext_vector_type(16))) __bf16 v16bf;
typedef __attribute__((ext_vector_type(8)))  float  v8f;

#define N_STATE 64
#define D_IN    128
#define D_OUT   128
#define BATCH   32
#define SEQ     8192
#define LCHUNK  64
#define NCHUNK  (SEQ / LCHUNK)            // 128
#define NCOLS_TOTAL (BATCH * NCHUNK)      // 4096
#define COLS    32                        // columns per workgroup
#define NWG     (NCOLS_TOTAL / COLS)      // 128
#define KDIM    256                       // GEMM K = 128 (Z) + 128 (U)
#define CT_TILES (COLS / 16)              // 2

// ---- workspace layout (float offsets) ----
constexpr size_t OFF_LAMRE = 0;
constexpr size_t OFF_LAMIM = 64;
constexpr size_t OFF_PIV   = 128;   // 2 floats (pivot reciprocal broadcast)
constexpr size_t OFF_FACRE = 192;   // 64
constexpr size_t OFF_FACIM = 256;   // 64
constexpr size_t OFF_MRE   = 320;                       // 64*64
constexpr size_t OFF_MIM   = OFF_MRE + 64 * 64;
constexpr size_t OFF_RRE   = OFF_MIM + 64 * 64;         // 64*192  [A_d | B_d] re
constexpr size_t OFF_RIM   = OFF_RRE + 64 * 192;        // 64*192
constexpr size_t OFF_P1RE  = OFF_RIM + 64 * 192;        // 64*64
constexpr size_t OFF_P1IM  = OFF_P1RE + 64 * 64;
constexpr size_t OFF_P2RE  = OFF_P1IM + 64 * 64;
constexpr size_t OFF_P2IM  = OFF_P2RE + 64 * 64;
constexpr size_t OFF_ML    = OFF_P2IM + 64 * 64;        // 128*128 f32
constexpr size_t OFF_E     = OFF_ML + 128 * 128;        // 4096*128 f32
constexpr size_t OFF_S     = OFF_E + (size_t)NCOLS_TOTAL * 128;
constexpr size_t OFF_GB    = OFF_S + (size_t)NCOLS_TOTAL * 128; // bf16[256*256]

// ---------------------------------------------------------------------------
// WMMA fragment loaders (CDNA5 16-bit layouts, CDNA5 ISA §7.12.2)
// ---------------------------------------------------------------------------
union FragU { uint4 q[2]; v16bf v; };

// A-matrix 16x32 bf16 fragment from row-major src (ld elements per row).
// lanes 0-15: M=r, K = k0+{0..7} then k0+{16..23}; lanes 16-31: +8 K offset.
__device__ inline v16bf load_a_frag(const __bf16* src, int ld, int row0, int k0) {
  int lane = threadIdx.x & 31;
  int half = lane >> 4;
  int r    = lane & 15;
  const __bf16* p = src + (size_t)(row0 + r) * ld + (k0 + half * 8);
  FragU f;
  f.q[0] = *(const uint4*)p;          // K = k0+half*8 .. +7
  f.q[1] = *(const uint4*)(p + 16);   // K = k0+16+half*8 .. +7
  return f.v;
}

// B-matrix 32x16 bf16 fragment from column-major storage srcT[col][row] (ld rows).
// lanes 0-15: N=r, K = k0+{0..7} then k0+{16..23}; lanes 16-31: +8 K offset.
__device__ inline v16bf load_b_frag(const __bf16* srcT, int ld, int col0, int k0) {
  int lane = threadIdx.x & 31;
  int half = lane >> 4;
  int r    = lane & 15;
  const __bf16* p = srcT + (size_t)(col0 + r) * ld + (k0 + half * 8);
  FragU f;
  f.q[0] = *(const uint4*)p;
  f.q[1] = *(const uint4*)(p + 16);
  return f.v;
}

// ---------------------------------------------------------------------------
// K1: setup — discretization, fused G matrix, A_d^64
// ---------------------------------------------------------------------------
__global__ __launch_bounds__(256)
void setup_kernel(const float* __restrict__ log_omega, const float* __restrict__ zeta_logit,
                  const float* __restrict__ Pm, const float* __restrict__ Qm,
                  const float* __restrict__ Bmat, const float* __restrict__ Cmat,
                  const float* __restrict__ Dmat, float* ws_, __bf16* __restrict__ Gb)
{
  volatile float* ws    = ws_;
  volatile float* lamre = ws + OFF_LAMRE;
  volatile float* lamim = ws + OFF_LAMIM;
  volatile float* piv   = ws + OFF_PIV;
  volatile float* facre = ws + OFF_FACRE;
  volatile float* facim = ws + OFF_FACIM;
  volatile float* Mre   = ws + OFF_MRE;
  volatile float* Mim   = ws + OFF_MIM;
  volatile float* Rre   = ws + OFF_RRE;
  volatile float* Rim   = ws + OFF_RIM;
  volatile float* P1re  = ws + OFF_P1RE;
  volatile float* P1im  = ws + OFF_P1IM;
  volatile float* P2re  = ws + OFF_P2RE;
  volatile float* P2im  = ws + OFF_P2IM;
  volatile float* ML    = ws + OFF_ML;

  const int tid = threadIdx.x;
  const float dt2 = 0.005f;      // DT/2
  const float sdt = 0.1f;        // sqrt(DT)

  // 1) eigenvalues
  if (tid < 64) {
    float om = __expf(log_omega[tid]);
    float z  = 1.f / (1.f + __expf(-zeta_logit[tid]));
    float t2 = fmaxf(1.f - z * z, 1e-8f);
    lamre[tid] = -z * om;
    lamim[tid] = om * sqrtf(t2);
  }
  __syncthreads();

  // 2) I_minus (M) and augmented RHS [I_plus | B*sqrt(DT)]
  for (int idx = tid; idx < 64 * 64; idx += 256) {
    int i = idx >> 6, j = idx & 63;
    float pq = 0.f;
    for (int k = 0; k < 4; ++k) pq += Pm[i * 4 + k] * Qm[j * 4 + k];
    float ar = pq + (i == j ? lamre[i] : 0.f);
    float ai = (i == j ? lamim[i] : 0.f);
    Mre[idx] = (i == j ? 1.f : 0.f) - dt2 * ar;
    Mim[idx] = -dt2 * ai;
    Rre[i * 192 + j] = (i == j ? 1.f : 0.f) + dt2 * ar;
    Rim[i * 192 + j] = dt2 * ai;
  }
  for (int idx = tid; idx < 64 * 128; idx += 256) {
    int i = idx >> 7, c = idx & 127;
    Rre[i * 192 + 64 + c] = Bmat[i * 128 + c] * sdt;
    Rim[i * 192 + 64 + c] = 0.f;
  }
  __syncthreads();

  // 3) complex Gauss-Jordan: M X = R  ->  R := [A_d | B_d]
  for (int k = 0; k < 64; ++k) {
    if (tid == 0) {
      float pr = Mre[k * 64 + k], pi = Mim[k * 64 + k];
      float d = pr * pr + pi * pi;
      piv[0] = pr / d; piv[1] = -pi / d;
    }
    __syncthreads();
    float ir = piv[0], ii = piv[1];
    if (tid < 64) {
      float a = Mre[k * 64 + tid], b = Mim[k * 64 + tid];
      Mre[k * 64 + tid] = a * ir - b * ii;
      Mim[k * 64 + tid] = a * ii + b * ir;
    }
    if (tid < 192) {
      float a = Rre[k * 192 + tid], b = Rim[k * 192 + tid];
      Rre[k * 192 + tid] = a * ir - b * ii;
      Rim[k * 192 + tid] = a * ii + b * ir;
    }
    __syncthreads();
    if (tid < 64) { facre[tid] = Mre[tid * 64 + k]; facim[tid] = Mim[tid * 64 + k]; }
    __syncthreads();
    for (int idx = tid; idx < 64 * 256; idx += 256) {
      int r = idx >> 8, c = idx & 255;
      if (r != k) {
        float fr = facre[r], fi = facim[r];
        if (c < 64) {
          float kr = Mre[k * 64 + c], ki = Mim[k * 64 + c];
          Mre[r * 64 + c] -= fr * kr - fi * ki;
          Mim[r * 64 + c] -= fr * ki + fi * kr;
        } else {
          int cc = c - 64;
          float kr = Rre[k * 192 + cc], ki = Rim[k * 192 + cc];
          Rre[r * 192 + cc] -= fr * kr - fi * ki;
          Rim[r * 192 + cc] -= fr * ki + fi * kr;
        }
      }
    }
    __syncthreads();
  }

  // 4) fused step matrix G (256x256 bf16), rows: [Z'(128); Y(128)], cols: [Z(128); U(128)]
  for (int idx = tid; idx < 64 * 64; idx += 256) {
    int i = idx >> 6, j = idx & 63;
    float ar = Rre[i * 192 + j], ai = Rim[i * 192 + j];
    Gb[(size_t)i * 256 + j]             = (__bf16)ar;
    Gb[(size_t)i * 256 + 64 + j]        = (__bf16)(-ai);
    Gb[(size_t)(64 + i) * 256 + j]      = (__bf16)ai;
    Gb[(size_t)(64 + i) * 256 + 64 + j] = (__bf16)ar;
  }
  for (int idx = tid; idx < 64 * 128; idx += 256) {
    int i = idx >> 7, c = idx & 127;
    Gb[(size_t)i * 256 + 128 + c]        = (__bf16)Rre[i * 192 + 64 + c];
    Gb[(size_t)(64 + i) * 256 + 128 + c] = (__bf16)Rim[i * 192 + 64 + c];
  }
  for (int idx = tid; idx < 128 * 64; idx += 256) {
    int o = idx >> 6, j = idx & 63;
    float sr = 0.f, si = 0.f;
    for (int n = 0; n < 64; ++n) {
      float cv = Cmat[o * 64 + n];
      sr += cv * Rre[n * 192 + j];
      si += cv * Rim[n * 192 + j];
    }
    Gb[(size_t)(128 + o) * 256 + j]      = (__bf16)sr;
    Gb[(size_t)(128 + o) * 256 + 64 + j] = (__bf16)(-si);
  }
  for (int idx = tid; idx < 128 * 128; idx += 256) {
    int o = idx >> 7, c = idx & 127;
    float s = Dmat[o * 128 + c];
    for (int n = 0; n < 64; ++n) s += Cmat[o * 64 + n] * Rre[n * 192 + 64 + c];
    Gb[(size_t)(128 + o) * 256 + 128 + c] = (__bf16)s;
  }

  // 5) A_L = A_d^64 by repeated squaring -> M_L (128x128 real block form)
  for (int idx = tid; idx < 64 * 64; idx += 256) {
    int i = idx >> 6, j = idx & 63;
    P1re[idx] = Rre[i * 192 + j];
    P1im[idx] = Rim[i * 192 + j];
  }
  __syncthreads();
  volatile float* sre = P1re; volatile float* sim = P1im;
  volatile float* dre = P2re; volatile float* dim = P2im;
  for (int s2 = 0; s2 < 6; ++s2) {
    for (int idx = tid; idx < 64 * 64; idx += 256) {
      int i = idx >> 6, j = idx & 63;
      float accr = 0.f, acci = 0.f;
      for (int k = 0; k < 64; ++k) {
        float ar = sre[i * 64 + k], ai = sim[i * 64 + k];
        float br = sre[k * 64 + j], bi = sim[k * 64 + j];
        accr += ar * br - ai * bi;
        acci += ar * bi + ai * br;
      }
      dre[idx] = accr; dim[idx] = acci;
    }
    __syncthreads();
    volatile float* t;
    t = sre; sre = dre; dre = t;
    t = sim; sim = dim; dim = t;
  }
  for (int idx = tid; idx < 64 * 64; idx += 256) {
    int i = idx >> 6, j = idx & 63;
    float ar = sre[idx], ai = sim[idx];
    ML[i * 128 + j]              = ar;
    ML[i * 128 + 64 + j]         = -ai;
    ML[(64 + i) * 128 + j]       = ai;
    ML[(64 + i) * 128 + 64 + j]  = ar;
  }
}

// ---------------------------------------------------------------------------
// K2/K4: chunked recurrence. FULL=false: local pass (store E).
//        FULL=true : final pass with carries (store y).
// ---------------------------------------------------------------------------
template<bool FULL>
__global__ __launch_bounds__(256)
void recur_kernel(const float* __restrict__ u, const __bf16* __restrict__ G,
                  const float* __restrict__ Sin, float* __restrict__ Eout,
                  float* __restrict__ y)
{
  __shared__ __align__(16) __bf16 rhsT[COLS * KDIM];  // [col][row], rows 0..127=Z, 128..255=U
  constexpr int H = FULL ? 2 : 1;
  const int tid   = threadIdx.x;
  const int wave  = tid >> 5;
  const int lane  = tid & 31;
  const int halfl = lane >> 4;
  const int r16   = lane & 15;
  const int b      = blockIdx.x >> 2;            // 4 wgs per batch
  const int chunk0 = (blockIdx.x & 3) * COLS;

  // G row-tile fragments resident in VGPRs for all 64 steps
  v16bf afrag[H][8];
#pragma unroll
  for (int h = 0; h < H; ++h)
#pragma unroll
    for (int kc = 0; kc < 8; ++kc)
      afrag[h][kc] = load_a_frag(G, KDIM, (wave + 8 * h) * 16, kc * 32);

  // init Z
  if constexpr (FULL) {
    for (int idx = tid; idx < COLS * 128; idx += 256) {
      int c = idx >> 7, row = idx & 127;
      size_t col = (size_t)b * NCHUNK + chunk0 + c;
      rhsT[c * KDIM + row] = (__bf16)Sin[col * 128 + row];
    }
  } else {
    for (int idx = tid; idx < COLS * 128; idx += 256) {
      int c = idx >> 7, row = idx & 127;
      rhsT[c * KDIM + row] = (__bf16)0.f;
    }
  }

  for (int j = 0; j < LCHUNK; ++j) {
    // stage U_j for all columns (coalesced f32 loads, bf16 into LDS)
    for (int idx = tid; idx < COLS * D_IN; idx += 256) {
      int c = idx >> 7, i = idx & 127;
      int t = (chunk0 + c) * LCHUNK + j;
      const float* up = u + ((size_t)b * SEQ + t) * D_IN + i;
      rhsT[c * KDIM + 128 + i] = (__bf16)(*up);
      __builtin_prefetch(up + D_IN, 0, 0);   // next step's element
    }
    __syncthreads();

    v8f acc[H][CT_TILES];
    const v8f vzero = {0.f, 0.f, 0.f, 0.f, 0.f, 0.f, 0.f, 0.f};
#pragma unroll
    for (int h = 0; h < H; ++h)
#pragma unroll
      for (int ct = 0; ct < CT_TILES; ++ct) acc[h][ct] = vzero;

#pragma unroll
    for (int ct = 0; ct < CT_TILES; ++ct) {
#pragma unroll
      for (int kc = 0; kc < 8; ++kc) {
        v16bf bf = load_b_frag(rhsT, KDIM, ct * 16, kc * 32);
#pragma unroll
        for (int h = 0; h < H; ++h)
          acc[h][ct] = __builtin_amdgcn_wmma_f32_16x16x32_bf16(
              false, afrag[h][kc], false, bf, (short)0, acc[h][ct], false, false);
      }
    }
    __syncthreads();

#pragma unroll
    for (int ct = 0; ct < CT_TILES; ++ct) {
      int c    = ct * 16 + r16;
      int row0 = wave * 16 + halfl * 8;       // 8 consecutive rows per lane
      // Z' -> LDS (bf16, packed 16B store)
      __align__(16) __bf16 zb[8];
#pragma unroll
      for (int m = 0; m < 8; ++m) zb[m] = (__bf16)acc[0][ct][m];
      *(uint4*)&rhsT[c * KDIM + row0] = *(const uint4*)zb;

      if constexpr (!FULL) {
        if (j == LCHUNK - 1) {
          size_t col = (size_t)b * NCHUNK + chunk0 + c;
          float4 e0 = make_float4(acc[0][ct][0], acc[0][ct][1], acc[0][ct][2], acc[0][ct][3]);
          float4 e1 = make_float4(acc[0][ct][4], acc[0][ct][5], acc[0][ct][6], acc[0][ct][7]);
          *(float4*)&Eout[col * 128 + row0]     = e0;
          *(float4*)&Eout[col * 128 + row0 + 4] = e1;
        }
      }
      if constexpr (FULL) {
        int t = (chunk0 + c) * LCHUNK + j;
        float* yp = y + ((size_t)b * SEQ + t) * D_OUT + row0;
        float4 y0 = make_float4(acc[1][ct][0], acc[1][ct][1], acc[1][ct][2], acc[1][ct][3]);
        float4 y1 = make_float4(acc[1][ct][4], acc[1][ct][5], acc[1][ct][6], acc[1][ct][7]);
        *(float4*)yp       = y0;
        *(float4*)(yp + 4) = y1;
      }
    }
    __syncthreads();
  }
}

// ---------------------------------------------------------------------------
// K3: carry scan across chunks (per batch): S[c+1] = M_L S[c] + E[c]
// ---------------------------------------------------------------------------
__global__ __launch_bounds__(128)
void carry_kernel(const float* __restrict__ E, const float* __restrict__ ML,
                  float* __restrict__ S)
{
  __shared__ float s[128];
  const int b = blockIdx.x;
  const int r = threadIdx.x;
  s[r] = 0.f;
  __syncthreads();
  for (int c = 0; c < NCHUNK; ++c) {
    S[((size_t)b * NCHUNK + c) * 128 + r] = s[r];
    float acc = E[((size_t)b * NCHUNK + c) * 128 + r];
    for (int k = 0; k < 128; ++k) acc += ML[r * 128 + k] * s[k];
    __syncthreads();
    s[r] = acc;
    __syncthreads();
  }
}

// ---------------------------------------------------------------------------
extern "C" void kernel_launch(void* const* d_in, const int* in_sizes, int n_in,
                              void* d_out, int out_size, void* d_ws, size_t ws_size,
                              hipStream_t stream) {
  const float* u          = (const float*)d_in[0];
  const float* log_omega  = (const float*)d_in[1];
  const float* zeta_logit = (const float*)d_in[2];
  const float* P          = (const float*)d_in[3];
  const float* Q          = (const float*)d_in[4];
  const float* Bm         = (const float*)d_in[5];
  const float* C          = (const float*)d_in[6];
  const float* D          = (const float*)d_in[7];
  float* y  = (float*)d_out;

  float*  ws = (float*)d_ws;
  float*  E  = ws + OFF_E;
  float*  S  = ws + OFF_S;
  float*  ML = ws + OFF_ML;
  __bf16* Gb = (__bf16*)(ws + OFF_GB);

  setup_kernel<<<1, 256, 0, stream>>>(log_omega, zeta_logit, P, Q, Bm, C, D, ws, Gb);
  recur_kernel<false><<<NWG, 256, 0, stream>>>(u, Gb, nullptr, E, nullptr);
  carry_kernel<<<BATCH, 128, 0, stream>>>(E, ML, S);
  recur_kernel<true><<<NWG, 256, 0, stream>>>(u, Gb, S, nullptr, y);
}
